// GRUExtracter_1511828488537
// MI455X (gfx1250) — compile-verified, matrix operand order checked by reading, not run on hardware
//
#include <hip/hip_runtime.h>
#include <hip/hip_bf16.h>

// ---------------------------------------------------------------------------
// Types for CDNA5 WMMA (wave32): v_wmma_f32_16x16x32_bf16
// ---------------------------------------------------------------------------
typedef __attribute__((ext_vector_type(16))) __bf16 v16bf;
typedef __attribute__((ext_vector_type(8)))  float  v8f;

#define TILE_M 128
#define TILE_N 128
#define TILE_K 32
#define LDSP   40   // 32 + 8 bf16 padding per row (80B, 16B aligned, conflict-free-ish)

// ---------------------------------------------------------------------------
// Generic bf16 WMMA GEMM: C[M,N] = A[M,K] * B[K,N] (+bias +residual, relu)
// A: bf16 row-major with leading dim lda. BT: bf16 [N][K] (B transposed, packed).
// C: optional fp32 out (ldc). C2: optional bf16 out (ldc2). R: optional fp32
// residual (ldr). M is blockIdx.y*128, N via blockIdx.x*128; all dims are
// multiples of tile sizes in this model (M=2048; N in {256,512,1536,2048};
// K in {64,256,512,1024,2048}).
// Block = 256 threads = 8 waves; wave grid 4(M)x2(N); wave tile 32x64 = 2x4 frags.
// ---------------------------------------------------------------------------
__global__ __launch_bounds__(256) void gemm_bf16_wmma(
    const __bf16* __restrict__ A, int lda,
    const __bf16* __restrict__ BT,
    float* __restrict__ C, int ldc,
    __bf16* __restrict__ C2, int ldc2,
    const float* __restrict__ bias,
    const float* __restrict__ R, int ldr,
    int K, int relu)
{
  __shared__ __bf16 sA[TILE_M * LDSP];
  __shared__ __bf16 sB[TILE_N * LDSP];

  const int tid  = threadIdx.x;
  const int lane = tid & 31;
  const int wave = tid >> 5;      // 0..7
  const int wm   = wave >> 1;     // 0..3 : 32-row strip
  const int wn   = wave & 1;      // 0..1 : 64-col strip

  const int rowBase = blockIdx.y * TILE_M;
  const int colBase = blockIdx.x * TILE_N;

  const int ldRow = tid >> 1;            // 0..127
  const int ldCol = (tid & 1) << 4;      // 0 or 16 (bf16 elems)

  v8f acc[2][4] = {};

  for (int k0 = 0; k0 < K; k0 += TILE_K) {
    // ---- global -> LDS (each thread: 16 bf16 of A tile + 16 bf16 of B tile)
    {
      const uint4* ga = (const uint4*)(A + (size_t)(rowBase + ldRow) * lda + k0 + ldCol);
      uint4 a0 = ga[0], a1 = ga[1];
      const uint4* gb = (const uint4*)(BT + (size_t)(colBase + ldRow) * K + k0 + ldCol);
      uint4 b0 = gb[0], b1 = gb[1];
      uint4* pa = (uint4*)&sA[ldRow * LDSP + ldCol];
      pa[0] = a0; pa[1] = a1;
      uint4* pb = (uint4*)&sB[ldRow * LDSP + ldCol];
      pb[0] = b0; pb[1] = b1;
    }
    __syncthreads();

    // ---- fragment loads in contiguous k order, same for A and B:
    // lane l: row = l%16, holds 16 contiguous k at half (l/16)*16
    const int kHalf = (lane >> 4) << 4;
    const int mrow  = lane & 15;

    union Frag { v16bf v; uint4 u[2]; };
    Frag af[2], bfr[4];
#pragma unroll
    for (int i = 0; i < 2; ++i) {
      const __bf16* p = &sA[(wm * 32 + i * 16 + mrow) * LDSP + kHalf];
      af[i].u[0] = *(const uint4*)p;
      af[i].u[1] = *(const uint4*)(p + 8);
    }
#pragma unroll
    for (int j = 0; j < 4; ++j) {
      const __bf16* p = &sB[(wn * 64 + j * 16 + mrow) * LDSP + kHalf];
      bfr[j].u[0] = *(const uint4*)p;
      bfr[j].u[1] = *(const uint4*)(p + 8);
    }

#pragma unroll
    for (int i = 0; i < 2; ++i)
#pragma unroll
      for (int j = 0; j < 4; ++j)
        acc[i][j] = __builtin_amdgcn_wmma_f32_16x16x32_bf16(
            false, af[i].v, false, bfr[j].v, (short)0, acc[i][j], false, false);

    __syncthreads();
  }

  // ---- epilogue: C/D layout: vgpr r, lane<16 -> M=r, lane>=16 -> M=8+r; N=lane%16
  const int halfm = lane >> 4;
  const int nloc  = lane & 15;
#pragma unroll
  for (int i = 0; i < 2; ++i) {
#pragma unroll
    for (int j = 0; j < 4; ++j) {
#pragma unroll
      for (int r = 0; r < 8; ++r) {
        int row = rowBase + wm * 32 + i * 16 + halfm * 8 + r;
        int col = colBase + wn * 64 + j * 16 + nloc;
        float v = acc[i][j][r];
        if (bias) v += bias[col];
        if (R)    v += R[(size_t)row * ldr + col];
        if (relu) v = v > 0.f ? v : 0.f;
        if (C)  C [(size_t)row * ldc  + col] = v;
        if (C2) C2[(size_t)row * ldc2 + col] = (__bf16)v;
      }
    }
  }
}

// ---------------------------------------------------------------------------
// GRU elementwise update:
// r = sig(gir+ghr+br); z = sig(giz+ghz+bz); n = tanh(gin + r*ghn + bn)
// h' = (1-z)*n + z*h.  gh/hprev may be null (step 0 => zeros).
// Writes fp32 h and bf16 into the layer output sequence (ld 8192, pre-offset).
// ---------------------------------------------------------------------------
__global__ __launch_bounds__(256) void gru_update(
    const float* __restrict__ gi, const float* __restrict__ gh,
    const float* __restrict__ b,  const float* __restrict__ hprev,
    float* __restrict__ h, __bf16* __restrict__ y)
{
  int idx = blockIdx.x * 256 + threadIdx.x;     // 2048*512 exact
  int n = idx >> 9;
  int j = idx & 511;
  size_t g = (size_t)n * 1536;
  float gir = gi[g + j], giz = gi[g + 512 + j], gin = gi[g + 1024 + j];
  float ghr = 0.f, ghz = 0.f, ghn = 0.f, hp = 0.f;
  if (gh)    { ghr = gh[g + j]; ghz = gh[g + 512 + j]; ghn = gh[g + 1024 + j]; }
  if (hprev) { hp = hprev[idx]; }
  float r  = 1.f / (1.f + expf(-(gir + ghr + b[j])));
  float z  = 1.f / (1.f + expf(-(giz + ghz + b[512 + j])));
  float nn = tanhf(gin + r * ghn + b[1024 + j]);
  float hn = (1.f - z) * nn + z * hp;
  h[idx] = hn;
  y[(size_t)n * 8192 + j] = (__bf16)hn;
}

// ---------------------------------------------------------------------------
__global__ void copy_bf16_tile(const __bf16* __restrict__ src, int lds_,
                               __bf16* __restrict__ dst, int ldd, int cols, int total)
{
  int idx = blockIdx.x * 256 + threadIdx.x;
  if (idx >= total) return;
  int n = idx / cols, c = idx % cols;
  dst[(size_t)n * ldd + c] = src[(size_t)n * lds_ + c];
}

__global__ void f32_to_bf16(const float* __restrict__ in, __bf16* __restrict__ out, int n)
{
  int i = blockIdx.x * 256 + threadIdx.x;
  if (i < n) out[i] = (__bf16)in[i];
}

// in: fp32 [K][N] row-major -> out: bf16 [N][K] (transposed, packed)
__global__ void f32_to_bf16_T(const float* __restrict__ in, __bf16* __restrict__ out,
                              int K, int N)
{
  int i = blockIdx.x * 256 + threadIdx.x;   // coalesced reads
  if (i < K * N) {
    int k = i / N, n = i % N;
    out[(size_t)n * K + k] = (__bf16)in[i];
  }
}

// layernorm over last dim 512, unit scale / zero bias. 1 wave32 per row.
__global__ __launch_bounds__(256) void layernorm512(const float* __restrict__ x,
                                                    float* __restrict__ out)
{
  int lane = threadIdx.x & 31;
  int wave = threadIdx.x >> 5;
  int row  = blockIdx.x * 8 + wave;
  const float* xr = x + (size_t)row * 512;
  float vals[16];
  float s = 0.f, s2 = 0.f;
#pragma unroll
  for (int i = 0; i < 16; ++i) {
    float v = xr[lane + i * 32];
    vals[i] = v; s += v; s2 += v * v;
  }
#pragma unroll
  for (int off = 16; off > 0; off >>= 1) {
    s  += __shfl_xor(s,  off, 32);
    s2 += __shfl_xor(s2, off, 32);
  }
  float mean = s * (1.f / 512.f);
  float var  = s2 * (1.f / 512.f) - mean * mean;
  float inv  = rsqrtf(var + 1e-5f);
  float* orow = out + (size_t)row * 512;
#pragma unroll
  for (int i = 0; i < 16; ++i)
    orow[lane + i * 32] = (vals[i] - mean) * inv;
}

// ---------------------------------------------------------------------------
extern "C" void kernel_launch(void* const* d_in, const int* in_sizes, int n_in,
                              void* d_out, int out_size, void* d_ws, size_t ws_size,
                              hipStream_t stream)
{
  (void)in_sizes; (void)n_in; (void)out_size; (void)ws_size;

  const int N = 2048;                 // B*neg
  const float* inp = (const float*)d_in[0];

  struct LW { const float* Wi; int Kin; const float* Wh; const float* b; };
  LW lw[4] = {
    { (const float*)d_in[1],  64,  (const float*)d_in[2],  (const float*)d_in[3]  },  // f0
    { (const float*)d_in[4],  512, (const float*)d_in[5],  (const float*)d_in[6]  },  // f1
    { (const float*)d_in[7],  64,  (const float*)d_in[8],  (const float*)d_in[9]  },  // b0
    { (const float*)d_in[10], 512, (const float*)d_in[11], (const float*)d_in[12] },  // b1
  };
  const float* fc_W   = (const float*)d_in[13];
  const float* fc_b   = (const float*)d_in[14];
  const float* bp1_W  = (const float*)d_in[15];
  const float* bp1_b  = (const float*)d_in[16];
  const float* bp2_W  = (const float*)d_in[17];
  const float* bp2_b  = (const float*)d_in[18];
  const float* ft1_W  = (const float*)d_in[19];
  const float* ft1_b  = (const float*)d_in[20];
  const float* ft2_W  = (const float*)d_in[21];
  const float* ft2_b  = (const float*)d_in[22];
  const float* ftallW = (const float*)d_in[23];
  const float* ftallb = (const float*)d_in[24];
  const float* ffn1W  = (const float*)d_in[25];
  const float* ffn2W  = (const float*)d_in[26];

  // ---- carve workspace -----------------------------------------------------
  char* wsb = (char*)d_ws;
  size_t off = 0;
  auto alloc = [&](size_t bytes) -> void* {
    void* p = wsb + off;
    off = (off + bytes + 255) & ~size_t(255);
    return p;
  };
  __bf16* xbf    = (__bf16*)alloc((size_t)N * 1024 * 2);
  __bf16* wWi[4], *wWh[4];
  for (int i = 0; i < 4; ++i) {
    wWi[i] = (__bf16*)alloc((size_t)lw[i].Kin * 1536 * 2);
    wWh[i] = (__bf16*)alloc((size_t)512 * 1536 * 2);
  }
  __bf16* wFc    = (__bf16*)alloc((size_t)512 * 1024 * 2);
  __bf16* wBp1   = (__bf16*)alloc((size_t)256 * 1024 * 2);
  __bf16* wBp2   = (__bf16*)alloc((size_t)512 * 256 * 2);
  __bf16* wFt1   = (__bf16*)alloc((size_t)256 * 1024 * 2);
  __bf16* wFt2   = (__bf16*)alloc((size_t)256 * 1024 * 2);
  __bf16* wFtall = (__bf16*)alloc((size_t)512 * 1024 * 2);
  __bf16* wFfn1  = (__bf16*)alloc((size_t)2048 * 512 * 2);
  __bf16* wFfn2  = (__bf16*)alloc((size_t)512 * 2048 * 2);
  __bf16* yA     = (__bf16*)alloc((size_t)N * 16 * 512 * 2);
  __bf16* yB     = (__bf16*)alloc((size_t)N * 16 * 512 * 2);
  float*  giBuf  = (float*) alloc((size_t)N * 1536 * 4);
  float*  ghBuf  = (float*) alloc((size_t)N * 1536 * 4);
  float*  hBuf   = (float*) alloc((size_t)N * 512 * 4);
  __bf16* lastBuf= (__bf16*)alloc((size_t)N * 1024 * 2);
  __bf16* catBuf = (__bf16*)alloc((size_t)N * 1024 * 2);
  __bf16* cat2   = (__bf16*)alloc((size_t)N * 1024 * 2);
  __bf16* bp1o   = (__bf16*)alloc((size_t)N * 256 * 2);
  float*  bpBuf  = (float*) alloc((size_t)N * 512 * 4);
  float*  ftpBuf = (float*) alloc((size_t)N * 512 * 4);
  float*  ftp2   = (float*) alloc((size_t)N * 512 * 4);
  __bf16* ftp2bf = (__bf16*)alloc((size_t)N * 512 * 2);
  __bf16* ffnh   = (__bf16*)alloc((size_t)N * 2048 * 2);
  float*  ffnOut = (float*) alloc((size_t)N * 512 * 4);

  // ---- weight / input conversion to bf16 (B matrices stored transposed) ----
  auto tconv = [&](const float* W, __bf16* out, int K, int Nc) {
    int total = K * Nc;
    f32_to_bf16_T<<<(total + 255) / 256, 256, 0, stream>>>(W, out, K, Nc);
  };
  {
    int total = N * 1024;
    f32_to_bf16<<<(total + 255) / 256, 256, 0, stream>>>(inp, xbf, total);
  }
  for (int i = 0; i < 4; ++i) {
    tconv(lw[i].Wi, wWi[i], lw[i].Kin, 1536);
    tconv(lw[i].Wh, wWh[i], 512, 1536);
  }
  tconv(fc_W,   wFc,    1024, 512);
  tconv(bp1_W,  wBp1,   1024, 256);
  tconv(bp2_W,  wBp2,   256,  512);
  tconv(ft1_W,  wFt1,   1024, 256);
  tconv(ft2_W,  wFt2,   1024, 256);
  tconv(ftallW, wFtall, 1024, 512);
  tconv(ffn1W,  wFfn1,  512,  2048);
  tconv(ffn2W,  wFfn2,  2048, 512);

  // ---- GEMM launcher (M fixed = 2048) --------------------------------------
  auto gemm = [&](const __bf16* A, int lda, const __bf16* BT,
                  float* C, int ldc, __bf16* C2, int ldc2,
                  const float* bias, const float* R, int ldr,
                  int Ncols, int K, int relu) {
    dim3 grid(Ncols / TILE_N, N / TILE_M);
    gemm_bf16_wmma<<<grid, 256, 0, stream>>>(A, lda, BT, C, ldc, C2, ldc2,
                                             bias, R, ldr, K, relu);
  };

  // ---- bidirectional 2-layer GRU -------------------------------------------
  for (int d = 0; d < 2; ++d) {
    for (int layer = 0; layer < 2; ++layer) {
      int li = d * 2 + layer;
      __bf16* yOut = (layer == 0) ? yA : yB;
      for (int s = 0; s < 16; ++s) {
        const __bf16* Aptr; int lda, Kin;
        if (layer == 0) {
          int t = d ? (15 - s) : s;           // backward dir reads flipped time
          Aptr = xbf + t * 64; lda = 1024; Kin = 64;
        } else {
          Aptr = yA + s * 512; lda = 8192; Kin = 512;
        }
        // gi = x_t @ Wi
        gemm(Aptr, lda, wWi[li], giBuf, 1536, nullptr, 0, nullptr, nullptr, 0,
             1536, Kin, 0);
        // gh = h_{s-1} @ Wh  (h0 == 0 -> skip)
        if (s > 0)
          gemm(yOut + (s - 1) * 512, 8192, wWh[li], ghBuf, 1536, nullptr, 0,
               nullptr, nullptr, 0, 1536, 512, 0);
        gru_update<<<(N * 512) / 256, 256, 0, stream>>>(
            giBuf, s ? ghBuf : nullptr, lw[li].b, s ? hBuf : nullptr,
            hBuf, yOut + s * 512);
      }
    }
    // last = concat(yf[:, -1, :], yb[:, 0, :])
    if (d == 0)
      copy_bf16_tile<<<(N * 512) / 256, 256, 0, stream>>>(
          yB + 15 * 512, 8192, lastBuf, 1024, 512, N * 512);
    else
      copy_bf16_tile<<<(N * 512) / 256, 256, 0, stream>>>(
          yB, 8192, lastBuf + 512, 1024, 512, N * 512);
  }

  // ---- dense tail ----------------------------------------------------------
  // sem = last @ fc_W + fc_b   (bf16 into cat[:, 0:512])
  gemm(lastBuf, 1024, wFc, nullptr, 0, catBuf, 1024, fc_b, nullptr, 0, 512, 1024, 0);
  // bp = (inputs @ bp1 + b) @ bp2 + b   (fp32 + bf16 into cat[:, 512:1024])
  gemm(xbf, 1024, wBp1, nullptr, 0, bp1o, 256, bp1_b, nullptr, 0, 256, 1024, 0);
  gemm(bp1o, 256, wBp2, bpBuf, 512, catBuf + 512, 1024, bp2_b, nullptr, 0, 512, 256, 0);
  // cat2[:, 0:512] = sem
  copy_bf16_tile<<<(N * 512) / 256, 256, 0, stream>>>(catBuf, 1024, cat2, 1024, 512, N * 512);
  // ftp = bp + concat(ft1, ft2)   (fp32 into ftpBuf; bf16 into cat2[:, 512:])
  gemm(catBuf, 1024, wFt1, ftpBuf, 512, cat2 + 512, 1024, ft1_b, bpBuf, 512, 256, 1024, 0);
  gemm(catBuf, 1024, wFt2, ftpBuf + 256, 512, cat2 + 768, 1024, ft2_b, bpBuf + 256, 512, 256, 1024, 0);
  // ftp2 = ftp + (cat2 @ ftall + b)
  gemm(cat2, 1024, wFtall, ftp2, 512, ftp2bf, 512, ftallb, ftpBuf, 512, 512, 1024, 0);
  // ffn hidden = relu(ftp2 @ ffn1)
  gemm(ftp2bf, 512, wFfn1, nullptr, 0, ffnh, 2048, nullptr, nullptr, 0, 2048, 512, 1);
  // ffnOut = ffn hidden @ ffn2 + ftp2 (residual)
  gemm(ffnh, 2048, wFfn2, ffnOut, 512, nullptr, 0, nullptr, ftp2, 512, 512, 2048, 0);
  // out = layernorm(ffnOut)
  layernorm512<<<N / 8, 256, 0, stream>>>(ffnOut, (float*)d_out);
}
